// SeqCondAttention_24154896073356
// MI455X (gfx1250) — matrix operand order repaired and checked by reading, not compiled, not run on hardware
//
#include <hip/hip_runtime.h>
#include <hip/hip_bf16.h>
#include <stdint.h>

// ---- problem constants ----
#define BQ 4
#define LQ 4096
#define DQ 1024
#define KQ 16
#define HQ 64
#define CKSQ 4
#define CCONV 2064
#define DIMMEM 1024
#define DIMMEMTOT 1040
#define MLQ (BQ*LQ)        // 16384 tokens
#define NCH 16             // scan chunks per (b,c)
#define CHL (LQ/NCH)       // 256

typedef __attribute__((ext_vector_type(16))) __bf16 v16bf;
typedef __attribute__((ext_vector_type(8)))  float  v8f;

static __device__ __forceinline__ uint16_t f2bf(float f){
  union{float f; uint32_t u;} x; x.f=f;
  uint32_t r = x.u + 0x7FFFu + ((x.u>>16)&1u);   // round-to-nearest-even
  return (uint16_t)(r>>16);
}
static __device__ __forceinline__ float sigmf(float x){ return 1.0f/(1.0f+__expf(-x)); }
static __device__ __forceinline__ float softplusf(float x){ return x>20.f ? x : log1pf(__expf(x)); }
static __device__ __forceinline__ float siluf(float x){ return x*sigmf(x); }

static __device__ __forceinline__ v16bf asfrag(uint4 lo, uint4 hi){
  union{ v16bf v; uint4 u[2]; } r; r.u[0]=lo; r.u[1]=hi; return r.v;
}

// ---------------- converts / packs ----------------
__global__ void cvt_f32_bf16(const float* __restrict__ in, uint16_t* __restrict__ out, int n){
  int i = blockIdx.x*blockDim.x + threadIdx.x;
  if (i < n) out[i] = f2bf(in[i]);
}

// Pack W[Kdim,N] fp32 row-major into bf16 WMMA B-fragment layout, column-padded to Npad:
// Wp[((kt*2+khalf)*Npad + n)*16 + e] = (n<N) ? W[kt*32 + khalf*8 + (e&7) + (e>>3)*16][n] : 0
__global__ void pack_w_bf16(const float* __restrict__ W, uint16_t* __restrict__ Wp,
                            int Kdim, int N, int Npad){
  int o = blockIdx.x*blockDim.x + threadIdx.x;
  int total = Kdim*Npad;
  if (o >= total) return;
  int e  = o & 15;
  int t  = o >> 4;
  int n  = t % Npad;
  int kh = t / Npad;                    // 0..Kdim/16-1
  int kk = (kh>>1)*32 + (kh&1)*8 + (e&7) + (e>>3)*16;  // source K row
  Wp[o] = (n < N) ? f2bf(W[(size_t)kk*N + n]) : (uint16_t)0;
}

// ---------------- WMMA fragment helpers (used by readout) ----------------
static __device__ __forceinline__ v16bf load_a(const uint16_t* A, int lda, int m0, int k0, int lane){
  int mrow = lane & 15;
  int kb = k0 + ((lane>>4)<<3);
  const uint16_t* p = A + (size_t)(m0+mrow)*lda + kb;
  union{ v16bf v; uint4 u[2]; } r;
  r.u[0] = *(const uint4*)p;
  r.u[1] = *(const uint4*)(p + 16);
  return r.v;
}
static __device__ __forceinline__ v16bf load_b(const uint16_t* Wp, int N, int n0, int kt, int lane){
  const uint16_t* p = Wp + (((size_t)(kt*2 + (lane>>4))*N + n0 + (lane&15)) << 4);
  union{ v16bf v; uint4 u[2]; } r;
  r.u[0] = ((const uint4*)p)[0];
  r.u[1] = ((const uint4*)p)[1];
  return r.v;
}

// ---------------- bf16 WMMA GEMM: C[Mtot,N] = A[Mtot,Kdim] @ W[Kdim,Npad] ----------------
// Wave computes 16x64 of C (4 accumulators). Block = 8 waves -> 32x256.
// Npad is a multiple of 256 (B pack is zero-padded) so there are NO conditionals
// around the WMMAs; only C stores are guarded by the true N.
// K loop is ping-pong double buffered (nkt must be even).
#define LOAD_KT(kt, aL,aH, q0L,q0H,q1L,q1H,q2L,q2H,q3L,q3H)            \
  { const uint16_t* p_ = ap + (size_t)(kt)*32;                         \
    aL = *(const uint4*)p_;  aH = *(const uint4*)(p_ + 16);            \
    const uint16_t* q_ = bp + (size_t)(kt)*bstep;                      \
    q0L = ((const uint4*)q_)[0];        q0H = ((const uint4*)q_)[1];   \
    q1L = ((const uint4*)(q_+256))[0];  q1H = ((const uint4*)(q_+256))[1]; \
    q2L = ((const uint4*)(q_+512))[0];  q2H = ((const uint4*)(q_+512))[1]; \
    q3L = ((const uint4*)(q_+768))[0];  q3H = ((const uint4*)(q_+768))[1]; }

#define MMA4(aL,aH, q0L,q0H,q1L,q1H,q2L,q2H,q3L,q3H)                                   \
  { v16bf a_ = asfrag(aL,aH);                                                          \
    acc0 = __builtin_amdgcn_wmma_f32_16x16x32_bf16(false,a_,false,asfrag(q0L,q0H),(short)0,acc0,false,false); \
    acc1 = __builtin_amdgcn_wmma_f32_16x16x32_bf16(false,a_,false,asfrag(q1L,q1H),(short)0,acc1,false,false); \
    acc2 = __builtin_amdgcn_wmma_f32_16x16x32_bf16(false,a_,false,asfrag(q2L,q2H),(short)0,acc2,false,false); \
    acc3 = __builtin_amdgcn_wmma_f32_16x16x32_bf16(false,a_,false,asfrag(q3L,q3H),(short)0,acc3,false,false); }

__global__ void gemm_bf16_wmma(const uint16_t* __restrict__ A, const uint16_t* __restrict__ Wp,
                               float* __restrict__ C, int N, int Npad, int Kdim,
                               int lda, int ldc){
  const int lane = threadIdx.x & 31;
  const int wave = threadIdx.x >> 5;
  const int mw = wave & 1;
  const int nw = wave >> 1;
  const int m0 = (blockIdx.x*2 + mw)*16;
  const int n0 = (blockIdx.y*4 + nw)*64;    // always < Npad (grid sized exactly)
  const int khalf = lane >> 4, nlane = lane & 15;

  const uint16_t* ap = A + (size_t)(m0 + nlane)*lda + (khalf<<3);
  const uint16_t* bp = Wp + (((size_t)khalf*Npad + n0 + nlane) << 4);
  const size_t bstep = ((size_t)2*Npad) << 4;

  v8f acc0 = {}, acc1 = {}, acc2 = {}, acc3 = {};
  const int nkt = Kdim >> 5;                // even for all call sites

  uint4 xaL,xaH, x0L,x0H,x1L,x1H,x2L,x2H,x3L,x3H;
  uint4 yaL,yaH, y0L,y0H,y1L,y1H,y2L,y2H,y3L,y3H;

  LOAD_KT(0, xaL,xaH, x0L,x0H,x1L,x1H,x2L,x2H,x3L,x3H);
  for (int kt = 0; kt + 2 <= nkt; kt += 2){
    LOAD_KT(kt+1, yaL,yaH, y0L,y0H,y1L,y1H,y2L,y2H,y3L,y3H);
    if (kt+3 < nkt) __builtin_prefetch(ap + (size_t)(kt+3)*32, 0, 1);
    MMA4(xaL,xaH, x0L,x0H,x1L,x1H,x2L,x2H,x3L,x3H);
    if (kt+2 < nkt){
      LOAD_KT(kt+2, xaL,xaH, x0L,x0H,x1L,x1H,x2L,x2H,x3L,x3H);
    }
    MMA4(yaL,yaH, y0L,y0H,y1L,y1H,y2L,y2H,y3L,y3H);
  }

  const int row = m0 + (khalf<<3);
  #define STORE_TILE(j, accv)                                              \
    if (n0 + (j)*16 < N){                                                  \
      float* cp_ = C + (size_t)row*ldc + n0 + (j)*16 + nlane;              \
      _Pragma("unroll")                                                    \
      for (int r = 0; r < 8; ++r) cp_[(size_t)r*ldc] = (accv)[r];          \
    }
  STORE_TILE(0, acc0)
  STORE_TILE(1, acc1)
  STORE_TILE(2, acc2)
  STORE_TILE(3, acc3)
  #undef STORE_TILE
}

// ---------------- conv + SiLU + phase/decay -> stack(re|im), qbuf ----------------
__global__ void conv_phase_kernel(const float* __restrict__ z, const float* __restrict__ conv_k,
    const float* __restrict__ theta_raw, const float* __restrict__ decay_slopes,
    const float* __restrict__ score_scale, const float* __restrict__ score_bias,
    const float* __restrict__ phase_scale,
    float* __restrict__ stack, float* __restrict__ qbuf)
{
  const int l = blockIdx.x;
  const int b = blockIdx.y;
  const int tid = threadIdx.x;
  __shared__ float pw_s[KQ];

  if (tid < KQ){
    int c = DIMMEM + tid;
    float acc = 0.f;
    #pragma unroll
    for (int j = 0; j < CKSQ; ++j){
      int lj = l - (CKSQ-1) + j;
      if (lj >= 0) acc += conv_k[j*CCONV + c] * z[((size_t)b*LQ + lj)*CCONV + c];
    }
    float s = siluf(acc);
    float slope = softplusf(decay_slopes[tid]);
    float dist = (float)(LQ-1 - l);
    float pw = softplusf(score_scale[tid]*s + score_bias[tid]) * __expf(-slope*dist);
    pw_s[tid] = fminf(fmaxf(pw, 1e-6f), 1000.0f);
  }
  __syncthreads();

  const size_t srow = ((size_t)b*LQ + l)*2048;
  for (int c = tid; c < DIMMEM; c += 256){
    float acc = 0.f;
    #pragma unroll
    for (int j = 0; j < CKSQ; ++j){
      int lj = l - (CKSQ-1) + j;
      if (lj >= 0) acc += conv_k[j*CCONV + c] * z[((size_t)b*LQ + lj)*CCONV + c];
    }
    float kv = siluf(acc);
    int k = c >> 6, h = c & 63;
    float th  = 0.001f + 2.999f*sigmf(theta_raw[k*HQ + h]);
    float ksc = kv*phase_scale[k];
    float phi = ksc/(1.0f + fabsf(ksc))*th;
    float kvw = kv*pw_s[k];
    float sn, cn; __sincosf(phi, &sn, &cn);
    stack[srow + c]          = kvw*cn;   // re
    stack[srow + DIMMEM + c] = kvw*sn;   // im
  }

  const size_t qrow = ((size_t)b*LQ + l)*1024;
  for (int c = tid; c < 1024; c += 256){
    int cz = DIMMEMTOT + c;
    float acc = 0.f;
    #pragma unroll
    for (int j = 0; j < CKSQ; ++j){
      int lj = l - (CKSQ-1) + j;
      if (lj >= 0) acc += conv_k[j*CCONV + cz] * z[((size_t)b*LQ + lj)*CCONV + cz];
    }
    qbuf[qrow + c] = siluf(acc);
  }
}

// ---------------- 3-phase chunked cumsum over L ----------------
__global__ void scan_p1(const float* __restrict__ stack, float* __restrict__ csum){
  int t = blockIdx.x*blockDim.x + threadIdx.x;
  if (t >= BQ*2048*NCH) return;
  int ch = t % NCH; int bc = t / NCH;
  int b = bc >> 11; int c = bc & 2047;
  size_t base = ((size_t)b*LQ + ch*CHL)*2048 + c;
  float s = 0.f;
  for (int i = 0; i < CHL; ++i) s += stack[base + (size_t)i*2048];
  csum[t] = s;
}
__global__ void scan_p2(float* __restrict__ csum){
  int bc = blockIdx.x*blockDim.x + threadIdx.x;
  if (bc >= BQ*2048) return;
  float run = 0.f;
  for (int ch = 0; ch < NCH; ++ch){
    float v = csum[bc*NCH + ch];
    csum[bc*NCH + ch] = run;           // exclusive prefix of chunk sums
    run += v;
  }
}
__global__ void scan_p3(float* __restrict__ stack, const float* __restrict__ csum){
  int t = blockIdx.x*blockDim.x + threadIdx.x;
  if (t >= BQ*2048*NCH) return;
  int ch = t % NCH; int bc = t / NCH;
  int b = bc >> 11; int c = bc & 2047;
  size_t base = ((size_t)b*LQ + ch*CHL)*2048 + c;
  float acc = csum[t];
  for (int i = 0; i < CHL; ++i){
    size_t idx = base + (size_t)i*2048;
    acc += stack[idx];
    stack[idx] = acc;
  }
}

// ---------------- combine: q-readout, oc, gate*silu, RMS-norm -> ybuf (bf16) ----------------
__global__ void combine_kernel(const float* __restrict__ stack, const float* __restrict__ qbuf,
    const float* __restrict__ gate, const float* __restrict__ w_int_raw,
    const float* __restrict__ norm_w, uint16_t* __restrict__ ybuf)
{
  const int l = blockIdx.x, b = blockIdx.y, tid = threadIdx.x;
  __shared__ float ysh[2048];
  __shared__ float red[256];
  const size_t srow = ((size_t)b*LQ + l)*2048;
  const size_t qrow = ((size_t)b*LQ + l)*1024;
  float ss = 0.f;
  const float scale = 0.125f;   // rsqrt(H=64)
  for (int c = tid; c < 2048; c += 256){
    int k   = c >> 7;
    int rem = c & 127;
    int h   = rem & 63;
    int isim= rem >> 6;
    int kq  = k >> 1;
    float re_acc = stack[srow + k*HQ + h];
    float im_acc = stack[srow + 1024 + k*HQ + h];
    float q_re = qbuf[qrow + kq*128 + h*2];
    float q_im = qbuf[qrow + kq*128 + h*2 + 1];
    float m = isim ? (im_acc*q_re - re_acc*q_im) : (re_acc*q_re + im_acc*q_im);
    m *= scale;
    float ew = __expf(w_int_raw[k*HQ + h]);
    float wint = ew/(ew + 1e-6f);
    float g = gate[srow + c];
    float yv = m*wint * g*sigmf(g);
    ysh[c] = yv;
    ss += yv*yv;
  }
  red[tid] = ss;
  __syncthreads();
  for (int off = 128; off > 0; off >>= 1){
    if (tid < off) red[tid] += red[tid + off];
    __syncthreads();
  }
  float rms = rsqrtf(red[0]*(1.0f/2048.0f) + 1e-6f);
  for (int c = tid; c < 2048; c += 256)
    ybuf[srow + c] = f2bf(ysh[c]*rms*norm_w[c]);
}

// ---------------- per-k readout GEMM with fused SwiGLU (WMMA) ----------------
__global__ void readout_wmma(const uint16_t* __restrict__ ybuf, const uint16_t* __restrict__ Wrp,
                             uint16_t* __restrict__ yact)
{
  const int lane = threadIdx.x & 31;
  const int wave = threadIdx.x >> 5;
  const int m0 = blockIdx.x * 16;
  const int nt = blockIdx.y*8 + wave;    // 12 val-tiles of 16 cols
  const int k  = blockIdx.z;
  if (nt >= 12) return;                  // whole-wave guard
  const uint16_t* A  = ybuf + k*128;     // lda = 2048
  const uint16_t* Wk = Wrp + (size_t)k*128*384;
  const int n0v = nt*16;
  const int n0g = 192 + nt*16;
  v8f cv = {}, cg = {};
  #pragma unroll
  for (int kt = 0; kt < 4; ++kt){
    v16bf a  = load_a(A, 2048, m0, kt*32, lane);
    v16bf bv = load_b(Wk, 384, n0v, kt, lane);
    v16bf bg = load_b(Wk, 384, n0g, kt, lane);
    cv = __builtin_amdgcn_wmma_f32_16x16x32_bf16(false, a, false, bv, (short)0, cv, false, false);
    cg = __builtin_amdgcn_wmma_f32_16x16x32_bf16(false, a, false, bg, (short)0, cg, false, false);
  }
  const int row = m0 + ((lane>>4)<<3);
  const int col = k*192 + n0v + (lane&15);
  #pragma unroll
  for (int r = 0; r < 8; ++r){
    float act = cv[r] * sigmf(cg[r]);
    yact[(size_t)(row+r)*3072 + col] = f2bf(act);
  }
}

// ---------------- launch ----------------
extern "C" void kernel_launch(void* const* d_in, const int* in_sizes, int n_in,
                              void* d_out, int out_size, void* d_ws, size_t ws_size,
                              hipStream_t stream)
{
  (void)in_sizes; (void)n_in; (void)out_size; (void)ws_size;
  const float* x            = (const float*)d_in[0];
  const float* W_in         = (const float*)d_in[1];
  const float* conv_k       = (const float*)d_in[2];
  const float* theta_raw    = (const float*)d_in[3];
  const float* w_int_raw    = (const float*)d_in[4];
  const float* decay_slopes = (const float*)d_in[5];
  const float* score_scale  = (const float*)d_in[6];
  const float* score_bias   = (const float*)d_in[7];
  const float* phase_scale  = (const float*)d_in[8];
  const float* W_gate       = (const float*)d_in[9];
  const float* norm_w       = (const float*)d_in[10];
  const float* W_readout    = (const float*)d_in[11];
  const float* W_out        = (const float*)d_in[12];
  float* out = (float*)d_out;

  const int NPAD_Z = 2304;   // 2064 rounded up to 256

  char* p = (char*)d_ws;
  auto alloc = [&](size_t bytes)->void*{
    void* r = (void*)p;
    p += (bytes + 255) & ~((size_t)255);
    return r;
  };
  uint16_t* xb    = (uint16_t*)alloc((size_t)MLQ*1024*2);
  uint16_t* winp  = (uint16_t*)alloc((size_t)1024*NPAD_Z*2);
  uint16_t* wgp   = (uint16_t*)alloc((size_t)1024*2048*2);
  uint16_t* wop   = (uint16_t*)alloc((size_t)3072*1024*2);
  uint16_t* wrp   = (uint16_t*)alloc((size_t)16*128*384*2);
  float*    z     = (float*)alloc((size_t)MLQ*2064*4);
  float*    stack = (float*)alloc((size_t)MLQ*2048*4);
  float*    qbuf  = (float*)alloc((size_t)MLQ*1024*4);
  float*    gate  = (float*)alloc((size_t)MLQ*2048*4);
  uint16_t* ybuf  = (uint16_t*)alloc((size_t)MLQ*2048*2);
  uint16_t* yact  = (uint16_t*)alloc((size_t)MLQ*3072*2);
  float*    csum  = (float*)alloc((size_t)BQ*2048*NCH*4);

  { int n = MLQ*1024; cvt_f32_bf16<<<(n+255)/256,256,0,stream>>>(x, xb, n); }
  { int n = 1024*NPAD_Z; pack_w_bf16<<<(n+255)/256,256,0,stream>>>(W_in,  winp, 1024, 2064, NPAD_Z); }
  { int n = 1024*2048;   pack_w_bf16<<<(n+255)/256,256,0,stream>>>(W_gate, wgp, 1024, 2048, 2048); }
  { int n = 3072*1024;   pack_w_bf16<<<(n+255)/256,256,0,stream>>>(W_out,  wop, 3072, 1024, 1024); }
  for (int k = 0; k < 16; ++k){
    int n = 128*384;
    pack_w_bf16<<<(n+255)/256,256,0,stream>>>(W_readout + (size_t)k*128*384,
                                              wrp + (size_t)k*128*384, 128, 384, 384);
  }

  { dim3 g(MLQ/32, NPAD_Z/256);
    gemm_bf16_wmma<<<g,256,0,stream>>>(xb, winp, z,    2064, NPAD_Z, 1024, 1024, 2064); }
  { dim3 g(MLQ/32, 2048/256);
    gemm_bf16_wmma<<<g,256,0,stream>>>(xb, wgp,  gate, 2048, 2048,   1024, 1024, 2048); }

  { dim3 g(LQ, BQ);
    conv_phase_kernel<<<g,256,0,stream>>>(z, conv_k, theta_raw, decay_slopes,
                                          score_scale, score_bias, phase_scale,
                                          stack, qbuf); }

  { int n = BQ*2048*NCH; scan_p1<<<(n+255)/256,256,0,stream>>>(stack, csum); }
  { int n = BQ*2048;     scan_p2<<<(n+255)/256,256,0,stream>>>(csum); }
  { int n = BQ*2048*NCH; scan_p3<<<(n+255)/256,256,0,stream>>>(stack, csum); }

  { dim3 g(LQ, BQ);
    combine_kernel<<<g,256,0,stream>>>(stack, qbuf, gate, w_int_raw, norm_w, ybuf); }

  { dim3 g(MLQ/16, 2, 16);
    readout_wmma<<<g,256,0,stream>>>(ybuf, wrp, yact); }

  { dim3 g(MLQ/32, 1024/256);
    gemm_bf16_wmma<<<g,256,0,stream>>>(yact, wop, out, 1024, 1024, 3072, 3072, 1024); }
}
